// TransMIL_14602888807007
// MI455X (gfx1250) — matrix operand order repaired and checked by reading, not compiled, not run on hardware
//
#include <hip/hip_runtime.h>
#include <hip/hip_bf16.h>
#include <math.h>

// ---------------- types ----------------
typedef __attribute__((ext_vector_type(16))) _Float16 v16h;
typedef __attribute__((ext_vector_type(8)))  _Float16 v8h;
typedef __attribute__((ext_vector_type(8)))  float    v8f;
typedef __attribute__((ext_vector_type(4)))  float    v4f;

// ---------------- model constants ----------------
#define N0      6000
#define INF_    2048
#define D1      1024
#define DMODEL  512
#define NT      6085    // 1 cls + 6084 tokens
#define ADD     84
#define HH      78
#define NP      6144
#define PAD     59
#define NHEAD   8
#define DH      64
#define MLAND   256
#define LFOLD   24

// ==========================================================================
// Generic batched WMMA GEMM:  C = cdiag*I + alpha*(A@B)  [+bias] [gelu]
//   A: M x K  (row stride lda, batch stride sA)
//   B: K x N  (row stride ldb) ... or N x K row-major when transB=1
//   C: M x N  (row stride ldc, batch stride sC)
// Requires K % 32 == 0.  Block tile 128x64, 8 waves (4 M x 2 N), each wave
// computes a 32x32 C tile with 4 WMMAs (2 LDS b128 loads per WMMA).
// Double-buffered LDS, branchless clamped global loads, f32->f16 in regs.
// ==========================================================================
__device__ __forceinline__ v16h load_afrag(const _Float16* as_, int row, int half_) {
  const _Float16* ap = as_ + row * 40 + half_ * 8;
  v8h lo = *(const v8h*)(ap);
  v8h hi = *(const v8h*)(ap + 16);
  return __builtin_shufflevector(lo, hi, 0,1,2,3,4,5,6,7,8,9,10,11,12,13,14,15);
}
__device__ __forceinline__ v16h load_bfrag(const _Float16* bs_, int row, int half_) {
  const _Float16* bp = bs_ + row * 40 + half_ * 8;
  v8h lo = *(const v8h*)(bp);
  v8h hi = *(const v8h*)(bp + 16);
  return __builtin_shufflevector(lo, hi, 0,1,2,3,4,5,6,7,8,9,10,11,12,13,14,15);
}

__global__ __launch_bounds__(256) void gemm_wmma(
    const float* __restrict__ A, const float* __restrict__ B,
    const float* __restrict__ bias, float* __restrict__ C,
    int M, int N, int K, int lda, int ldb, int ldc,
    long long sA, long long sB, long long sC,
    int transB, int fuse, float alpha, float cdiag)
{
  __shared__ __align__(16) _Float16 As[2][128 * 40];
  __shared__ __align__(16) _Float16 Bs[2][64 * 40];

  const int tid  = threadIdx.x;
  const int lane = tid & 31;
  const int wave = tid >> 5;
  const int wm   = wave & 3;   // M sub-tile (32 rows each)
  const int wn   = wave >> 2;  // N sub-tile (32 cols each)
  const int bm   = blockIdx.y * 128;
  const int bn   = blockIdx.x * 64;
  const int bat  = blockIdx.z;

  A += (long long)bat * sA;
  B += (long long)bat * sB;
  C += (long long)bat * sC;

  // ---- staging index maps ----
  // A: thread -> (row = tid>>1 in 0..127, q = tid&1), 16 floats at k = q*16..
  const int  sa_row  = tid >> 1;
  const int  sa_q    = tid & 1;
  const int  sa_grow = bm + sa_row;
  const bool sa_ok   = sa_grow < M;
  const long long sa_base =
      (long long)(sa_ok ? sa_grow : (M - 1)) * lda + sa_q * 16;

  // B: transB -> (row = tid>>2 in 0..63, q = tid&3), 8 floats at k = q*8..
  //    NN     -> (tk = tid>>6 in 0..3, nn = tid&63), k = tk*8 + r, col = nn.
  const int  sbt_row = tid >> 2;
  const int  sbt_q   = tid & 3;
  const int  sb_tk   = tid >> 6;
  const int  sb_nn   = tid & 63;
  bool sb_ok;
  long long sb_base;
  if (transB) {
    int gn  = bn + sbt_row;
    sb_ok   = gn < N;
    sb_base = (long long)(sb_ok ? gn : (N - 1)) * ldb + sbt_q * 8;
  } else {
    int gn  = bn + sb_nn;
    sb_ok   = gn < N;
    sb_base = (long long)(sb_tk * 8) * ldb + (sb_ok ? gn : (N - 1));
  }

  v8f acc00 = {}, acc01 = {}, acc10 = {}, acc11 = {};
  const int m16   = lane & 15;
  const int half_ = lane >> 4;

  float ar[16], br[8];
  const int nk = K >> 5;

  // ---- prologue: load + stage tile 0 into buffer 0 ----
  {
    #pragma unroll
    for (int u = 0; u < 4; ++u) {
      v4f f = *(const v4f*)(A + sa_base + u * 4);
      #pragma unroll
      for (int e = 0; e < 4; ++e) ar[u * 4 + e] = f[e];
    }
    if (transB) {
      v4f g0 = *(const v4f*)(B + sb_base);
      v4f g1 = *(const v4f*)(B + sb_base + 4);
      #pragma unroll
      for (int e = 0; e < 4; ++e) { br[e] = g0[e]; br[4 + e] = g1[e]; }
    } else {
      #pragma unroll
      for (int r = 0; r < 8; ++r) br[r] = B[sb_base + (long long)r * ldb];
    }
    v8h ha0, ha1, hb;
    #pragma unroll
    for (int e = 0; e < 8; ++e) {
      ha0[e] = (_Float16)(sa_ok ? ar[e] : 0.0f);
      ha1[e] = (_Float16)(sa_ok ? ar[8 + e] : 0.0f);
      hb[e]  = (_Float16)(sb_ok ? br[e] : 0.0f);
    }
    *(v8h*)(As[0] + sa_row * 40 + sa_q * 16)     = ha0;
    *(v8h*)(As[0] + sa_row * 40 + sa_q * 16 + 8) = ha1;
    if (transB) *(v8h*)(Bs[0] + sbt_row * 40 + sbt_q * 8) = hb;
    else        *(v8h*)(Bs[0] + sb_nn * 40 + sb_tk * 8)   = hb;
  }

  for (int t = 0; t < nk; ++t) {
    __syncthreads();
    const int  cur  = t & 1;
    const bool more = (t + 1) < nk;

    // ---- issue next tile's global loads (land during the WMMAs) ----
    if (more) {
      const int k1 = (t + 1) * 32;
      #pragma unroll
      for (int u = 0; u < 4; ++u) {
        v4f f = *(const v4f*)(A + sa_base + k1 + u * 4);
        #pragma unroll
        for (int e = 0; e < 4; ++e) ar[u * 4 + e] = f[e];
      }
      if (transB) {
        v4f g0 = *(const v4f*)(B + sb_base + k1);
        v4f g1 = *(const v4f*)(B + sb_base + k1 + 4);
        #pragma unroll
        for (int e = 0; e < 4; ++e) { br[e] = g0[e]; br[4 + e] = g1[e]; }
      } else {
        const long long kb = sb_base + (long long)k1 * ldb;
        #pragma unroll
        for (int r = 0; r < 8; ++r) br[r] = B[kb + (long long)r * ldb];
      }
      if ((t + 2) < nk) {   // prefetch tile after next into caches
        const int k2 = (t + 2) * 32;
        __builtin_prefetch(A + sa_base + k2, 0, 1);
        if (transB) __builtin_prefetch(B + sb_base + k2, 0, 1);
        else        __builtin_prefetch(B + sb_base + (long long)k2 * ldb, 0, 1);
      }
    }

    // ---- compute from current LDS buffer: 8 frag loads, 4 WMMAs ----
    const _Float16* as_ = As[cur];
    const _Float16* bs_ = Bs[cur];
    v16h a0 = load_afrag(as_, wm * 32 + m16,       half_);
    v16h a1 = load_afrag(as_, wm * 32 + 16 + m16,  half_);
    v16h b0 = load_bfrag(bs_, wn * 32 + m16,       half_);
    v16h b1 = load_bfrag(bs_, wn * 32 + 16 + m16,  half_);
    acc00 = __builtin_amdgcn_wmma_f32_16x16x32_f16(false, a0, false, b0,
                                                   (short)0, acc00, false, false);
    acc01 = __builtin_amdgcn_wmma_f32_16x16x32_f16(false, a0, false, b1,
                                                   (short)0, acc01, false, false);
    acc10 = __builtin_amdgcn_wmma_f32_16x16x32_f16(false, a1, false, b0,
                                                   (short)0, acc10, false, false);
    acc11 = __builtin_amdgcn_wmma_f32_16x16x32_f16(false, a1, false, b1,
                                                   (short)0, acc11, false, false);

    // ---- stage next tile into the other buffer ----
    if (more) {
      const int nxt = 1 - cur;
      v8h ha0, ha1, hb;
      #pragma unroll
      for (int e = 0; e < 8; ++e) {
        ha0[e] = (_Float16)(sa_ok ? ar[e] : 0.0f);
        ha1[e] = (_Float16)(sa_ok ? ar[8 + e] : 0.0f);
        hb[e]  = (_Float16)(sb_ok ? br[e] : 0.0f);
      }
      *(v8h*)(As[nxt] + sa_row * 40 + sa_q * 16)     = ha0;
      *(v8h*)(As[nxt] + sa_row * 40 + sa_q * 16 + 8) = ha1;
      if (transB) *(v8h*)(Bs[nxt] + sbt_row * 40 + sbt_q * 8) = hb;
      else        *(v8h*)(Bs[nxt] + sb_nn * 40 + sb_tk * 8)   = hb;
    }
  }

  // ---- epilogue: C layout lane n = lane%16, VGPR r -> m = r + (lane/16)*8 ----
  #pragma unroll
  for (int i = 0; i < 2; ++i) {
    #pragma unroll
    for (int j = 0; j < 2; ++j) {
      v8f acc = i ? (j ? acc11 : acc10) : (j ? acc01 : acc00);
      int gcol = bn + wn * 32 + j * 16 + m16;
      if (gcol >= N) continue;
      #pragma unroll
      for (int r = 0; r < 8; ++r) {
        int grow = bm + wm * 32 + i * 16 + r + half_ * 8;
        if (grow >= M) continue;
        float v = alpha * acc[r];
        if (grow == gcol) v += cdiag;
        if (fuse & 1) v += bias[gcol];
        if (fuse & 2) v = 0.5f * v * (1.0f + erff(v * 0.70710678118654752f));
        C[(long long)grow * ldc + gcol] = v;
      }
    }
  }
}

// ---------------- block reductions ----------------
__device__ __forceinline__ float blk_sum(float v, float* red) {
  int t = threadIdx.x;
  red[t] = v; __syncthreads();
  for (int s = blockDim.x >> 1; s > 0; s >>= 1) {
    if (t < s) red[t] += red[t + s];
    __syncthreads();
  }
  float r = red[0]; __syncthreads();
  return r;
}
__device__ __forceinline__ float blk_max(float v, float* red) {
  int t = threadIdx.x;
  red[t] = v; __syncthreads();
  for (int s = blockDim.x >> 1; s > 0; s >>= 1) {
    if (t < s) red[t] = fmaxf(red[t], red[t + s]);
    __syncthreads();
  }
  float r = red[0]; __syncthreads();
  return r;
}

// ---------------- layernorm (one block per row) ----------------
__global__ __launch_bounds__(256) void layernorm_k(
    const float* __restrict__ in, float* __restrict__ out,
    const float* __restrict__ g, const float* __restrict__ b, int ncols)
{
  __shared__ float red[256];
  long long row = blockIdx.x;
  const float* x = in + row * ncols;
  float* y = out + row * ncols;
  float s = 0.0f;
  for (int c = threadIdx.x; c < ncols; c += 256) s += x[c];
  float mu = blk_sum(s, red) / (float)ncols;
  float q = 0.0f;
  for (int c = threadIdx.x; c < ncols; c += 256) { float d = x[c] - mu; q += d * d; }
  float var = blk_sum(q, red) / (float)ncols;
  float rstd = rsqrtf(var + 1e-5f);
  for (int c = threadIdx.x; c < ncols; c += 256)
    y[c] = (x[c] - mu) * rstd * g[c] + b[c];
}

// ---------------- softmax along last dim (one block per row) ----------------
__global__ __launch_bounds__(256) void softmax_k(float* __restrict__ data, int ncols)
{
  __shared__ float red[256];
  long long row = blockIdx.x;
  float* x = data + row * (long long)ncols;
  float m = -3.0e38f;
  for (int c = threadIdx.x; c < ncols; c += 256) m = fmaxf(m, x[c]);
  m = blk_max(m, red);
  float s = 0.0f;
  for (int c = threadIdx.x; c < ncols; c += 256) { float e = expf(x[c] - m); x[c] = e; s += e; }
  s = blk_sum(s, red);
  float inv = 1.0f / s;
  for (int c = threadIdx.x; c < ncols; c += 256) x[c] *= inv;
}

// ---------------- misc elementwise kernels ----------------
__global__ void build_h_k(const float* __restrict__ h2, const float* __restrict__ cls,
                          float* __restrict__ h)
{
  int row = blockIdx.x, c = threadIdx.x;
  float v;
  if (row == 0)            v = cls[c];
  else if (row <= N0)      v = h2[(long long)(row - 1) * DMODEL + c];
  else                     v = h2[(long long)(row - 1 - N0) * DMODEL + c];
  h[(long long)row * DMODEL + c] = v;
}

__global__ void zero_rows_k(float* __restrict__ p) {
  p[(long long)blockIdx.x * DMODEL + threadIdx.x] = 0.0f;
}

__global__ void landmark_k(const float* __restrict__ qkv, float* __restrict__ out,
                           int coloff, float scale)
{
  int idx = blockIdx.x * blockDim.x + threadIdx.x;       // 8*256*64
  if (idx >= NHEAD * MLAND * DH) return;
  int d = idx & 63, j = (idx >> 6) & 255, h = idx >> 14;
  const float* base = qkv + (long long)(j * LFOLD) * (3 * DMODEL) + coloff + h * DH + d;
  float s = 0.0f;
  for (int li = 0; li < LFOLD; ++li) s += base[(long long)li * (3 * DMODEL)];
  out[idx] = s * (scale / (float)LFOLD);
}

__global__ __launch_bounds__(256) void absrowsum_k(const float* __restrict__ a2,
                                                   float* __restrict__ out) {
  __shared__ float red[256];
  long long row = blockIdx.x;                            // h*256+i
  float v = fabsf(a2[row * MLAND + threadIdx.x]);
  float s = blk_sum(v, red);
  if (threadIdx.x == 0) out[row] = s;
}
__global__ __launch_bounds__(256) void abscolsum_k(const float* __restrict__ a2,
                                                   float* __restrict__ out) {
  __shared__ float red[256];
  int h = blockIdx.x >> 8, j = blockIdx.x & 255;
  float v = fabsf(a2[(long long)h * MLAND * MLAND + (long long)threadIdx.x * MLAND + j]);
  float s = blk_sum(v, red);
  if (threadIdx.x == 0) out[blockIdx.x] = s;
}
__global__ __launch_bounds__(256) void maxmul_k(const float* __restrict__ rs,
                                                const float* __restrict__ cs,
                                                float* __restrict__ denom) {
  __shared__ float red[256];
  float m1 = -3.0e38f, m2 = -3.0e38f;
  for (int i = threadIdx.x; i < NHEAD * MLAND; i += 256) {
    m1 = fmaxf(m1, rs[i]); m2 = fmaxf(m2, cs[i]);
  }
  m1 = blk_max(m1, red);
  m2 = blk_max(m2, red);
  if (threadIdx.x == 0) denom[0] = m1 * m2;
}
__global__ void zinit_k(const float* __restrict__ a2, const float* __restrict__ denom,
                        float* __restrict__ z) {
  int h = blockIdx.x >> 8, j = blockIdx.x & 255, i = threadIdx.x;
  float d = denom[0];
  z[(long long)blockIdx.x * MLAND + i] =
      a2[(long long)h * MLAND * MLAND + (long long)i * MLAND + j] / d;
}
__global__ void diag_axpy_k(const float* __restrict__ in, float* __restrict__ out,
                            float cdiag, float scale) {
  long long idx = (long long)blockIdx.x * 256 + threadIdx.x;  // 8*256*256
  int r = (int)((idx >> 8) & 255), c = (int)(idx & 255);
  out[idx] = scale * in[idx] + ((r == c) ? cdiag : 0.0f);
}

// residual depthwise conv along sequence (kernel 33, pad 16), add into merged
__global__ __launch_bounds__(512) void resconv_k(const float* __restrict__ qkv,
                                                 const float* __restrict__ w,
                                                 float* __restrict__ merged) {
  int i = blockIdx.x;
  int h = threadIdx.x >> 6, d = threadIdx.x & 63;
  float s = 0.0f;
  #pragma unroll
  for (int t = 0; t < 33; ++t) {
    int src = i + t - 16;
    if (src >= 0 && src < NP)
      s += w[h * 33 + t] * qkv[(long long)src * (3 * DMODEL) + 2 * DMODEL + h * DH + d];
  }
  merged[(long long)i * DMODEL + threadIdx.x] += s;
}

__global__ __launch_bounds__(512) void resadd_k(float* __restrict__ h,
                                                const float* __restrict__ proj) {
  long long row = blockIdx.x;
  h[row * DMODEL + threadIdx.x] += proj[(row + PAD) * DMODEL + threadIdx.x];
}

__global__ __launch_bounds__(512) void copyrow0_k(const float* __restrict__ src,
                                                  float* __restrict__ dst) {
  dst[threadIdx.x] = src[threadIdx.x];
}

// fused identity + dw7 + dw5 + dw3 mixer on 78x78 grid, channels-last tokens
__global__ __launch_bounds__(512) void cnn_k(
    const float* __restrict__ hin, float* __restrict__ hout,
    const float* __restrict__ w7, const float* __restrict__ b7,
    const float* __restrict__ w5, const float* __restrict__ b5,
    const float* __restrict__ w3, const float* __restrict__ b3)
{
  int x = blockIdx.x, y = blockIdx.y, c = threadIdx.x;
  const float* feat = hin + DMODEL;   // skip cls row
  float acc = b7[c] + b5[c] + b3[c];
  #pragma unroll
  for (int dy = -3; dy <= 3; ++dy) {
    #pragma unroll
    for (int dx = -3; dx <= 3; ++dx) {
      float wv = w7[c * 49 + (dy + 3) * 7 + (dx + 3)];
      if (dy >= -2 && dy <= 2 && dx >= -2 && dx <= 2)
        wv += w5[c * 25 + (dy + 2) * 5 + (dx + 2)];
      if (dy >= -1 && dy <= 1 && dx >= -1 && dx <= 1)
        wv += w3[c * 9 + (dy + 1) * 3 + (dx + 1)];
      if (dy == 0 && dx == 0) wv += 1.0f;    // identity term
      int yy = y + dy, xx = x + dx;
      if (yy >= 0 && yy < HH && xx >= 0 && xx < HH)
        acc += wv * feat[(long long)(yy * HH + xx) * DMODEL + c];
    }
  }
  hout[(long long)(1 + y * HH + x) * DMODEL + c] = acc;
}

// final layernorm(row 0) + 512->2 classifier
__global__ __launch_bounds__(512) void head_k(
    const float* __restrict__ h, const float* __restrict__ g,
    const float* __restrict__ b, const float* __restrict__ fcw,
    const float* __restrict__ fcb, float* __restrict__ out)
{
  __shared__ float red[512];
  __shared__ float ln[512];
  int t = threadIdx.x;
  float v = h[t];
  red[t] = v; __syncthreads();
  for (int s = 256; s > 0; s >>= 1) { if (t < s) red[t] += red[t + s]; __syncthreads(); }
  float mu = red[0] / 512.0f; __syncthreads();
  float d = v - mu;
  red[t] = d * d; __syncthreads();
  for (int s = 256; s > 0; s >>= 1) { if (t < s) red[t] += red[t + s]; __syncthreads(); }
  float rstd = rsqrtf(red[0] / 512.0f + 1e-5f); __syncthreads();
  ln[t] = d * rstd * g[t] + b[t];
  __syncthreads();
  for (int k = 0; k < 2; ++k) {
    red[t] = ln[t] * fcw[t * 2 + k]; __syncthreads();
    for (int s = 256; s > 0; s >>= 1) { if (t < s) red[t] += red[t + s]; __syncthreads(); }
    if (t == 0) out[k] = red[0] + fcb[k];
    __syncthreads();
  }
}

// ---------------- host-side helpers ----------------
static inline void launch_gemm(const float* A, const float* B, const float* bias, float* C,
                               int M, int N, int K, int lda, int ldb, int ldc,
                               long long sA, long long sB, long long sC, int batch,
                               int transB, int fuse, float alpha, float cdiag,
                               hipStream_t st)
{
  dim3 g((N + 63) / 64, (M + 127) / 128, batch);
  gemm_wmma<<<g, 256, 0, st>>>(A, B, bias, C, M, N, K, lda, ldb, ldc,
                               sA, sB, sC, transB, fuse, alpha, cdiag);
}

struct AttnWS {
  float *xp, *qkv, *s1, *s2, *s3, *ql, *kl;
  float *z, *z2, *xz, *w1, *w2;
  float *a3v, *az, *merged, *rsum, *csum, *denom;
};

static void attention_layer(float* hbuf, const float* lng, const float* lnb,
                            const float* qkvw, const float* outw, const float* outb,
                            const float* resw, const AttnWS& w, hipStream_t st)
{
  const long long HM2 = (long long)MLAND * MLAND;       // 65536
  const long long HS1 = (long long)NP * MLAND;          // per-head s1/s3
  // x_ln -> xp (front-padded with PAD zero rows)
  zero_rows_k<<<PAD, DMODEL, 0, st>>>(w.xp);
  layernorm_k<<<NT, 256, 0, st>>>(hbuf, w.xp + (long long)PAD * DMODEL, lng, lnb, DMODEL);
  // qkv = xp @ qkv_w                         (6144x512)@(512x1536)
  launch_gemm(w.xp, qkvw, nullptr, w.qkv, NP, 3 * DMODEL, DMODEL,
              DMODEL, 3 * DMODEL, 3 * DMODEL, 0, 0, 0, 1, 0, 0, 1.0f, 0.0f, st);
  // landmarks (q scaled by DH^-1/2)
  landmark_k<<<(NHEAD * MLAND * DH + 255) / 256, 256, 0, st>>>(w.qkv, w.ql, 0, 0.125f);
  landmark_k<<<(NHEAD * MLAND * DH + 255) / 256, 256, 0, st>>>(w.qkv, w.kl, DMODEL, 1.0f);
  // s1 = (q * scale) @ kl^T  per head       (6144x64)@(64x256)
  launch_gemm(w.qkv, w.kl, nullptr, w.s1, NP, MLAND, DH,
              3 * DMODEL, DH, MLAND, DH, (long long)MLAND * DH, HS1, NHEAD,
              1, 0, 0.125f, 0.0f, st);
  // s2 = ql @ kl^T                          (256x64)@(64x256)
  launch_gemm(w.ql, w.kl, nullptr, w.s2, MLAND, MLAND, DH,
              DH, DH, MLAND, (long long)MLAND * DH, (long long)MLAND * DH, HM2, NHEAD,
              1, 0, 1.0f, 0.0f, st);
  // s3 = ql @ k^T                           (256x64)@(64x6144)
  launch_gemm(w.ql, w.qkv + DMODEL, nullptr, w.s3, MLAND, NP, DH,
              DH, 3 * DMODEL, NP, (long long)MLAND * DH, DH, (long long)MLAND * NP, NHEAD,
              1, 0, 1.0f, 0.0f, st);
  // softmaxes (in-place)
  softmax_k<<<NHEAD * NP, 256, 0, st>>>(w.s1, MLAND);
  softmax_k<<<NHEAD * MLAND, 256, 0, st>>>(w.s2, MLAND);
  softmax_k<<<NHEAD * MLAND, 256, 0, st>>>(w.s3, NP);
  // Moore-Penrose pinv of a2 (=s2)
  absrowsum_k<<<NHEAD * MLAND, 256, 0, st>>>(w.s2, w.rsum);
  abscolsum_k<<<NHEAD * MLAND, 256, 0, st>>>(w.s2, w.csum);
  maxmul_k<<<1, 256, 0, st>>>(w.rsum, w.csum, w.denom);
  zinit_k<<<NHEAD * MLAND, 256, 0, st>>>(w.s2, w.denom, w.z);
  float* zin = w.z; float* zout = w.z2;
  for (int it = 0; it < 6; ++it) {
    // xz = a2 @ z
    launch_gemm(w.s2, zin, nullptr, w.xz, MLAND, MLAND, MLAND,
                MLAND, MLAND, MLAND, HM2, HM2, HM2, NHEAD, 0, 0, 1.0f, 0.0f, st);
    // w1 = 7I - xz
    diag_axpy_k<<<NHEAD * MLAND, 256, 0, st>>>(w.xz, w.w1, 7.0f, -1.0f);
    // w2 = 15I - xz @ w1
    launch_gemm(w.xz, w.w1, nullptr, w.w2, MLAND, MLAND, MLAND,
                MLAND, MLAND, MLAND, HM2, HM2, HM2, NHEAD, 0, 0, -1.0f, 15.0f, st);
    // w3(=w1) = 13I - xz @ w2
    launch_gemm(w.xz, w.w2, nullptr, w.w1, MLAND, MLAND, MLAND,
                MLAND, MLAND, MLAND, HM2, HM2, HM2, NHEAD, 0, 0, -1.0f, 13.0f, st);
    // z' = 0.25 * z @ w3
    launch_gemm(zin, w.w1, nullptr, zout, MLAND, MLAND, MLAND,
                MLAND, MLAND, MLAND, HM2, HM2, HM2, NHEAD, 0, 0, 0.25f, 0.0f, st);
    float* tmp = zin; zin = zout; zout = tmp;
  }
  // a3v = a3 @ v                            (256x6144)@(6144x64)
  launch_gemm(w.s3, w.qkv + 2 * DMODEL, nullptr, w.a3v, MLAND, DH, NP,
              NP, 3 * DMODEL, DH, (long long)MLAND * NP, DH, (long long)MLAND * DH, NHEAD,
              0, 0, 1.0f, 0.0f, st);
  // az = a2_inv @ a3v                       (256x256)@(256x64)
  launch_gemm(zin, w.a3v, nullptr, w.az, MLAND, DH, MLAND,
              MLAND, DH, DH, HM2, (long long)MLAND * DH, (long long)MLAND * DH, NHEAD,
              0, 0, 1.0f, 0.0f, st);
  // merged[i][h*64+d] = a1 @ az             (6144x256)@(256x64), per-head col slice
  launch_gemm(w.s1, w.az, nullptr, w.merged, NP, DH, MLAND,
              MLAND, DH, DMODEL, HS1, (long long)MLAND * DH, DH, NHEAD,
              0, 0, 1.0f, 0.0f, st);
  // += depthwise sequence conv on v
  resconv_k<<<NP, 512, 0, st>>>(w.qkv, resw, w.merged);
  // out projection (write into xp, which is free now)
  launch_gemm(w.merged, outw, outb, w.xp, NP, DMODEL, DMODEL,
              DMODEL, DMODEL, DMODEL, 0, 0, 0, 1, 0, 1, 1.0f, 0.0f, st);
  // residual: h += proj[PAD:]
  resadd_k<<<NT, 512, 0, st>>>(hbuf, w.xp);
}

// ---------------- entry point ----------------
extern "C" void kernel_launch(void* const* d_in, const int* in_sizes, int n_in,
                              void* d_out, int out_size, void* d_ws, size_t ws_size,
                              hipStream_t stream) {
  (void)in_sizes; (void)n_in; (void)out_size; (void)ws_size;
  // input order: x, then params in dict insertion order
  const float* x       = (const float*)d_in[0];
  const float* fc1_w1  = (const float*)d_in[1];
  const float* fc1_b1  = (const float*)d_in[2];
  const float* fc1_g   = (const float*)d_in[3];
  const float* fc1_bL  = (const float*)d_in[4];
  const float* fc1_w2  = (const float*)d_in[5];
  const float* fc1_b2  = (const float*)d_in[6];
  const float* cls_tok = (const float*)d_in[7];
  const float* norm_g  = (const float*)d_in[8];
  const float* norm_b  = (const float*)d_in[9];
  const float* fc_w    = (const float*)d_in[10];
  const float* fc_b    = (const float*)d_in[11];
  const float* pw7     = (const float*)d_in[12];
  const float* pb7     = (const float*)d_in[13];
  const float* pw5     = (const float*)d_in[14];
  const float* pb5     = (const float*)d_in[15];
  const float* pw3     = (const float*)d_in[16];
  const float* pb3     = (const float*)d_in[17];
  const float* l1_ln_g = (const float*)d_in[18];
  const float* l1_ln_b = (const float*)d_in[19];
  const float* l1_qkvw = (const float*)d_in[20];
  const float* l1_outw = (const float*)d_in[21];
  const float* l1_outb = (const float*)d_in[22];
  const float* l1_resw = (const float*)d_in[23];
  const float* l2_ln_g = (const float*)d_in[24];
  const float* l2_ln_b = (const float*)d_in[25];
  const float* l2_qkvw = (const float*)d_in[26];
  const float* l2_outw = (const float*)d_in[27];
  const float* l2_outb = (const float*)d_in[28];
  const float* l2_resw = (const float*)d_in[29];

  float* ws = (float*)d_ws;
  size_t o = 0;
  auto alloc = [&](size_t n) { float* p = ws + o; o += (n + 15) & ~(size_t)15; return p; };

  float* h1    = alloc((size_t)N0 * D1);
  float* h2    = alloc((size_t)N0 * DMODEL);
  float* hbuf  = alloc((size_t)NT * DMODEL);
  float* hbuf2 = alloc((size_t)NT * DMODEL);

  AttnWS aw;
  aw.xp     = alloc((size_t)NP * DMODEL);
  aw.qkv    = alloc((size_t)NP * 3 * DMODEL);
  aw.s1     = alloc((size_t)NHEAD * NP * MLAND);
  aw.s3     = alloc((size_t)NHEAD * MLAND * NP);
  aw.merged = alloc((size_t)NP * DMODEL);
  aw.ql     = alloc((size_t)NHEAD * MLAND * DH);
  aw.kl     = alloc((size_t)NHEAD * MLAND * DH);
  aw.a3v    = alloc((size_t)NHEAD * MLAND * DH);
  aw.az     = alloc((size_t)NHEAD * MLAND * DH);
  aw.s2     = alloc((size_t)NHEAD * MLAND * MLAND);
  aw.z      = alloc((size_t)NHEAD * MLAND * MLAND);
  aw.z2     = alloc((size_t)NHEAD * MLAND * MLAND);
  aw.xz     = alloc((size_t)NHEAD * MLAND * MLAND);
  aw.w1     = alloc((size_t)NHEAD * MLAND * MLAND);
  aw.w2     = alloc((size_t)NHEAD * MLAND * MLAND);
  aw.rsum   = alloc(NHEAD * MLAND);
  aw.csum   = alloc(NHEAD * MLAND);
  aw.denom  = alloc(16);

  // ---- MLP front-end ----
  // h1 = gelu(x @ fc1_w1 + b1)
  launch_gemm(x, fc1_w1, fc1_b1, h1, N0, D1, INF_, INF_, D1, D1,
              0, 0, 0, 1, 0, 3, 1.0f, 0.0f, stream);
  // h1 = layernorm(h1) (in place)
  layernorm_k<<<N0, 256, 0, stream>>>(h1, h1, fc1_g, fc1_bL, D1);
  // h2 = gelu(h1 @ fc1_w2 + b2)
  launch_gemm(h1, fc1_w2, fc1_b2, h2, N0, DMODEL, D1, D1, DMODEL, DMODEL,
              0, 0, 0, 1, 0, 3, 1.0f, 0.0f, stream);
  // h = [cls; h2; h2[:ADD]]
  build_h_k<<<NT, DMODEL, 0, stream>>>(h2, cls_tok, hbuf);

  // ---- layer 1 attention (residual inside) ----
  attention_layer(hbuf, l1_ln_g, l1_ln_b, l1_qkvw, l1_outw, l1_outb, l1_resw, aw, stream);

  // ---- conv mixer on the 78x78 token grid ----
  copyrow0_k<<<1, DMODEL, 0, stream>>>(hbuf, hbuf2);
  {
    dim3 g(HH, HH);
    cnn_k<<<g, DMODEL, 0, stream>>>(hbuf, hbuf2, pw7, pb7, pw5, pb5, pw3, pb3);
  }

  // ---- layer 2 attention ----
  attention_layer(hbuf2, l2_ln_g, l2_ln_b, l2_qkvw, l2_outw, l2_outb, l2_resw, aw, stream);

  // ---- final layernorm(row0) + classifier ----
  head_k<<<1, 512, 0, stream>>>(hbuf2, norm_g, norm_b, fc_w, fc_b, (float*)d_out);
}
